// Attention_2920577761368
// MI455X (gfx1250) — matrix operand order repaired and checked by reading, not compiled, not run on hardware
//
#include <hip/hip_runtime.h>
#include <hip/hip_bf16.h>

#define BB    4
#define NNN   2048
#define DDIM  1024
#define HEADS 16
#define DH    64
#define ROT   32

typedef __attribute__((ext_vector_type(16))) __bf16 v16bf;
typedef __attribute__((ext_vector_type(8)))  __bf16 v8bf;
typedef __attribute__((ext_vector_type(8)))  float  v8f;

// ---------------------------------------------------------------------------
// WMMA helper (gfx1250, wave32). D = A(16x32 bf16) x B(32x16 bf16) + C(f32)
// ---------------------------------------------------------------------------
static __device__ __forceinline__ v8f wmma_bf16(v16bf a, v16bf b, v8f c) {
  return __builtin_amdgcn_wmma_f32_16x16x32_bf16(
      /*neg_a=*/false, a, /*neg_b=*/false, b,
      /*c_mod=*/(short)0, c, /*reuse_a=*/false, /*reuse_b=*/false);
}

static __device__ __forceinline__ v16bf combine8(v8bf lo, v8bf hi) {
  v16bf r;
#pragma unroll
  for (int i = 0; i < 8; ++i) { r[i] = lo[i]; r[8 + i] = hi[i]; }
  return r;
}

// A fragment (ISA 16-bit A 16x32 layout): lane 0-15 -> row, halves hold
// K = kh..kh+7 and kh+16..kh+23 with kh = (lane/16)*8. Source K-contiguous.
static __device__ __forceinline__ v16bf load_frag_a(const __bf16* base, int ld) {
  int lane = threadIdx.x & 31;
  int r  = lane & 15;
  int kh = (lane >> 4) << 3;
  const __bf16* p = base + (long)r * ld + kh;
  v8bf lo = *(const v8bf*)p;
  v8bf hi = *(const v8bf*)(p + 16);
  return combine8(lo, hi);
}

// B fragment (matches A layout): lane 0-15 -> col, halves hold
// K = kb..kb+7 and kb+16..kb+23 with kb = (lane/16)*8. Source stores
// B^T (col-major => K-contiguous rows of length ld).
static __device__ __forceinline__ v16bf load_frag_b(const __bf16* base, int ld) {
  int lane = threadIdx.x & 31;
  int c  = lane & 15;
  int kb = (lane >> 4) << 3;   // 0 or 8
  const __bf16* p = base + (long)c * ld + kb;
  v8bf lo = *(const v8bf*)p;
  v8bf hi = *(const v8bf*)(p + 16);
  return combine8(lo, hi);
}

// A fragment from f32 source (convert on the fly) — used for P in PV.
static __device__ __forceinline__ v16bf load_frag_a_f32(const float* base, int ld) {
  int lane = threadIdx.x & 31;
  int r  = lane & 15;
  int kh = (lane >> 4) << 3;
  const float* p = base + (long)r * ld + kh;
  v16bf a;
#pragma unroll
  for (int i = 0; i < 8; ++i) {
    a[i]     = (__bf16)p[i];
    a[8 + i] = (__bf16)p[16 + i];
  }
  return a;
}

// ---------------------------------------------------------------------------
// 1) f32 -> bf16 convert, convert+transpose ([K][N] -> [N][K]) for weights,
//    and bf16 transpose for V ([bh][n][dh] -> [bh][dh][n]).
// ---------------------------------------------------------------------------
__global__ void cvt_bf16_kernel(const float* __restrict__ src,
                                __bf16* __restrict__ dst, int n) {
  int i = blockIdx.x * blockDim.x + threadIdx.x;
  if (i < n) dst[i] = (__bf16)src[i];
}

__global__ void cvt_transpose_kernel(const float* __restrict__ src,
                                     __bf16* __restrict__ dst, int K, int N) {
  int k = blockIdx.x * blockDim.x + threadIdx.x;
  int n = blockIdx.y;
  if (k < K) dst[(long)n * K + k] = (__bf16)src[(long)k * N + n];
}

__global__ void v_transpose_kernel(const __bf16* __restrict__ V,   // [bh][n][dh]
                                   __bf16* __restrict__ Vt) {      // [bh][dh][n]
  int n  = blockIdx.x * blockDim.x + threadIdx.x;   // 0..NNN-1
  int d  = blockIdx.y;                              // 0..DH-1
  int bh = blockIdx.z;                              // 0..B*H-1
  Vt[((long)bh * DH + d) * NNN + n] = V[((long)bh * NNN + n) * DH + d];
}

// ---------------------------------------------------------------------------
// 2) Projection GEMM: C[8192][1024] = Xbf[M][K] * Wt[N][K], scatter bf16
//    into [b][h][n][dh]. 256 threads (8 waves), wave tile 32x32, block tile
//    64(M) x 128(N). A tile staged through LDS.
// ---------------------------------------------------------------------------
__global__ void proj_gemm_kernel(const __bf16* __restrict__ A,   // [M][DDIM]
                                 const __bf16* __restrict__ Bt,  // [DDIM][DDIM] n-major
                                 __bf16* __restrict__ Out)       // [b][h][n][dh]
{
  __shared__ __bf16 As[64 * 32];
  const int K = DDIM;
  int wave = threadIdx.x >> 5;
  int wr = wave & 1, wc = wave >> 1;
  int row0 = blockIdx.y * 64 + wr * 32;
  int col0 = blockIdx.x * 128 + wc * 32;

  int t  = threadIdx.x;
  int lr = t >> 2;            // 0..63 : row within block tile
  int lk = (t & 3) << 3;      // 0,8,16,24
  const __bf16* Arow = A + (long)(blockIdx.y * 64 + lr) * K + lk;

  v8f acc[2][2] = {};
  for (int k0 = 0; k0 < K; k0 += 32) {
    __syncthreads();
    *(v8bf*)(As + lr * 32 + lk) = *(const v8bf*)(Arow + k0);
    __builtin_prefetch(Arow + k0 + 64, 0, 0);
    __syncthreads();

    v16bf a0 = load_frag_a(As + (wr * 32 + 0)  * 32, 32);
    v16bf a1 = load_frag_a(As + (wr * 32 + 16) * 32, 32);
    v16bf b0 = load_frag_b(Bt + (long)(col0 + 0)  * K + k0, K);
    v16bf b1 = load_frag_b(Bt + (long)(col0 + 16) * K + k0, K);
    acc[0][0] = wmma_bf16(a0, b0, acc[0][0]);
    acc[0][1] = wmma_bf16(a0, b1, acc[0][1]);
    acc[1][0] = wmma_bf16(a1, b0, acc[1][0]);
    acc[1][1] = wmma_bf16(a1, b1, acc[1][1]);
  }

  int lane = t & 31;
  int cc = lane & 15, rh = lane >> 4;
#pragma unroll
  for (int mi = 0; mi < 2; ++mi)
#pragma unroll
    for (int ni = 0; ni < 2; ++ni)
#pragma unroll
      for (int i = 0; i < 8; ++i) {
        int row = row0 + mi * 16 + rh * 8 + i;
        int col = col0 + ni * 16 + cc;
        int b = row / NNN, n = row % NNN;
        int h = col / DH,  d = col % DH;
        Out[(((long)(b * HEADS + h)) * NNN + n) * DH + d] = (__bf16)acc[mi][ni][i];
      }
}

// ---------------------------------------------------------------------------
// 3) Partial RoPE, in place on bf16 [b][h][n][dh] (first ROT dims of head)
// ---------------------------------------------------------------------------
__global__ void rope_kernel(__bf16* __restrict__ T, const float* __restrict__ rope) {
  long idx   = (long)blockIdx.x * blockDim.x + threadIdx.x;
  long total = (long)BB * HEADS * NNN * (ROT / 2);
  if (idx >= total) return;
  int  d   = (int)(idx & 15);
  long rem = idx >> 4;
  int  n   = (int)(rem % NNN);
  long bh  = rem / NNN;
  __bf16* p = T + (bh * NNN + n) * DH;
  float f = rope[(long)n * ROT + d];   // freqs duplicated: f[d+16] == f[d]
  float c = cosf(f), s = sinf(f);
  float t0 = (float)p[d];
  float t1 = (float)p[d + 16];
  p[d]      = (__bf16)(t0 * c - t1 * s);
  p[d + 16] = (__bf16)(t1 * c + t0 * s);
}

// ---------------------------------------------------------------------------
// 4) dots = Q K^T * scale per (b,h). Q,K stored [bh][n][dh] (dh-contiguous);
//    K rows serve directly as the (K^T)-transposed B operand.
// ---------------------------------------------------------------------------
__global__ void dots_kernel(const __bf16* __restrict__ Q,
                            const __bf16* __restrict__ Kmat,
                            float* __restrict__ dots) {
  int z = blockIdx.z;
  const __bf16* Qb = Q    + (long)z * NNN * DH;
  const __bf16* Kb = Kmat + (long)z * NNN * DH;
  float*        Db = dots + (long)z * NNN * NNN;

  int wave = threadIdx.x >> 5;
  int wr = wave & 1, wc = wave >> 1;
  int row0 = blockIdx.y * 64 + wr * 32;
  int col0 = blockIdx.x * 128 + wc * 32;

  v8f acc[2][2] = {};
#pragma unroll
  for (int k0 = 0; k0 < DH; k0 += 32) {
    v16bf a0 = load_frag_a(Qb + (long)(row0 + 0)  * DH + k0, DH);
    v16bf a1 = load_frag_a(Qb + (long)(row0 + 16) * DH + k0, DH);
    v16bf b0 = load_frag_b(Kb + (long)(col0 + 0)  * DH + k0, DH);
    v16bf b1 = load_frag_b(Kb + (long)(col0 + 16) * DH + k0, DH);
    acc[0][0] = wmma_bf16(a0, b0, acc[0][0]);
    acc[0][1] = wmma_bf16(a0, b1, acc[0][1]);
    acc[1][0] = wmma_bf16(a1, b0, acc[1][0]);
    acc[1][1] = wmma_bf16(a1, b1, acc[1][1]);
  }

  const float scale = 0.125f;  // dh^-0.5, dh = 64
  int lane = threadIdx.x & 31;
  int cc = lane & 15, rh = lane >> 4;
#pragma unroll
  for (int mi = 0; mi < 2; ++mi)
#pragma unroll
    for (int ni = 0; ni < 2; ++ni)
#pragma unroll
      for (int i = 0; i < 8; ++i) {
        int row = row0 + mi * 16 + rh * 8 + i;
        int col = col0 + ni * 16 + cc;
        Db[(long)row * NNN + col] = acc[mi][ni][i] * scale;
      }
}

// ---------------------------------------------------------------------------
// 5) Row softmax in place (one block of 256 threads per row of 2048)
// ---------------------------------------------------------------------------
__global__ void softmax_kernel(float* __restrict__ dots) {
  float* p = dots + (long)blockIdx.x * NNN;
  __shared__ float red[8];
  __shared__ float smax, ssum;
  int t = threadIdx.x, lane = t & 31, w = t >> 5;

  float m = -1e30f;
  for (int j = t; j < NNN; j += 256) m = fmaxf(m, p[j]);
#pragma unroll
  for (int off = 16; off; off >>= 1) m = fmaxf(m, __shfl_xor(m, off));
  if (lane == 0) red[w] = m;
  __syncthreads();
  if (t == 0) {
    float mm = red[0];
#pragma unroll
    for (int i = 1; i < 8; ++i) mm = fmaxf(mm, red[i]);
    smax = mm;
  }
  __syncthreads();
  m = smax;

  float s = 0.0f;
  for (int j = t; j < NNN; j += 256) {
    float e = __expf(p[j] - m);
    p[j] = e;
    s += e;
  }
#pragma unroll
  for (int off = 16; off; off >>= 1) s += __shfl_xor(s, off);
  if (lane == 0) red[w] = s;
  __syncthreads();
  if (t == 0) {
    float ss = 0.0f;
#pragma unroll
    for (int i = 0; i < 8; ++i) ss += red[i];
    ssum = ss;
  }
  __syncthreads();
  float inv = 1.0f / ssum;
  for (int j = t; j < NNN; j += 256) p[j] *= inv;
}

// ---------------------------------------------------------------------------
// 6) O = P V per (b,h). P is f32 (converted per-fragment); Vt[bh][dh][n] is
//    the pre-transposed V so B fragments are K-contiguous vector loads.
//    Block tile 128 (rows) x 64 (cols=dh); scatter bf16 to [b*n][DIM].
// ---------------------------------------------------------------------------
__global__ void pv_kernel(const float* __restrict__ P,
                          const __bf16* __restrict__ Vt,  // [bh][dh][n]
                          __bf16* __restrict__ Obf) {
  int z = blockIdx.z;
  int b = z / HEADS, h = z % HEADS;
  const float*  Pb  = P  + (long)z * NNN * NNN;
  const __bf16* Vtb = Vt + (long)z * DH * NNN;

  int wave = threadIdx.x >> 5;
  int wr = wave & 3, wc = wave >> 2;          // 4 x 2 waves
  int row0 = blockIdx.y * 128 + wr * 32;
  int col0 = wc * 32;

  v8f acc[2][2] = {};
  for (int k0 = 0; k0 < NNN; k0 += 32) {
    v16bf a0 = load_frag_a_f32(Pb + (long)(row0 + 0)  * NNN + k0, NNN);
    v16bf a1 = load_frag_a_f32(Pb + (long)(row0 + 16) * NNN + k0, NNN);
    v16bf b0 = load_frag_b(Vtb + (long)(col0 + 0)  * NNN + k0, NNN);
    v16bf b1 = load_frag_b(Vtb + (long)(col0 + 16) * NNN + k0, NNN);
    acc[0][0] = wmma_bf16(a0, b0, acc[0][0]);
    acc[0][1] = wmma_bf16(a0, b1, acc[0][1]);
    acc[1][0] = wmma_bf16(a1, b0, acc[1][0]);
    acc[1][1] = wmma_bf16(a1, b1, acc[1][1]);
  }

  int lane = threadIdx.x & 31;
  int cc = lane & 15, rh = lane >> 4;
#pragma unroll
  for (int mi = 0; mi < 2; ++mi)
#pragma unroll
    for (int ni = 0; ni < 2; ++ni)
#pragma unroll
      for (int i = 0; i < 8; ++i) {
        int n = row0 + mi * 16 + rh * 8 + i;
        int d = col0 + ni * 16 + cc;
        Obf[((long)(b * NNN + n)) * DDIM + h * DH + d] = (__bf16)acc[mi][ni][i];
      }
}

// ---------------------------------------------------------------------------
// 7) out = Obf @ Wo^T(nk) + bo + x   (f32 output to d_out)
// ---------------------------------------------------------------------------
__global__ void out_gemm_kernel(const __bf16* __restrict__ A,   // [M][DDIM]
                                const __bf16* __restrict__ Bt,  // [DDIM][DDIM] n-major
                                const float* __restrict__ bo,
                                const float* __restrict__ x,
                                float* __restrict__ out) {
  const int K = DDIM;
  int wave = threadIdx.x >> 5;
  int wr = wave & 1, wc = wave >> 1;
  int row0 = blockIdx.y * 64 + wr * 32;
  int col0 = blockIdx.x * 128 + wc * 32;

  v8f acc[2][2] = {};
  for (int k0 = 0; k0 < K; k0 += 32) {
    v16bf a0 = load_frag_a(A + (long)(row0 + 0)  * K + k0, K);
    v16bf a1 = load_frag_a(A + (long)(row0 + 16) * K + k0, K);
    v16bf b0 = load_frag_b(Bt + (long)(col0 + 0)  * K + k0, K);
    v16bf b1 = load_frag_b(Bt + (long)(col0 + 16) * K + k0, K);
    acc[0][0] = wmma_bf16(a0, b0, acc[0][0]);
    acc[0][1] = wmma_bf16(a0, b1, acc[0][1]);
    acc[1][0] = wmma_bf16(a1, b0, acc[1][0]);
    acc[1][1] = wmma_bf16(a1, b1, acc[1][1]);
  }

  int lane = threadIdx.x & 31;
  int cc = lane & 15, rh = lane >> 4;
#pragma unroll
  for (int mi = 0; mi < 2; ++mi)
#pragma unroll
    for (int ni = 0; ni < 2; ++ni)
#pragma unroll
      for (int i = 0; i < 8; ++i) {
        int row = row0 + mi * 16 + rh * 8 + i;
        int col = col0 + ni * 16 + cc;
        long idx = (long)row * DDIM + col;
        out[idx] = acc[mi][ni][i] + bo[col] + x[idx];
      }
}

// ---------------------------------------------------------------------------
// Host-side orchestration
// ---------------------------------------------------------------------------
extern "C" void kernel_launch(void* const* d_in, const int* in_sizes, int n_in,
                              void* d_out, int out_size, void* d_ws, size_t ws_size,
                              hipStream_t stream) {
  const float* x    = (const float*)d_in[0];
  const float* rope = (const float*)d_in[1];
  const float* Wq   = (const float*)d_in[2];
  const float* Wk   = (const float*)d_in[3];
  const float* Wv   = (const float*)d_in[4];
  const float* Wo   = (const float*)d_in[5];
  const float* bo   = (const float*)d_in[6];

  const long M     = (long)BB * NNN;        // 8192
  const long XEL   = M * DDIM;              // 8,388,608
  const long WEL   = (long)DDIM * DDIM;     // 1,048,576

  char* ws = (char*)d_ws;
  size_t off = 0;
  __bf16* Xbf = (__bf16*)(ws + off); off += XEL * 2;
  __bf16* Wqt = (__bf16*)(ws + off); off += WEL * 2;
  __bf16* Wkt = (__bf16*)(ws + off); off += WEL * 2;
  __bf16* Wvt = (__bf16*)(ws + off); off += WEL * 2;
  __bf16* Wot = (__bf16*)(ws + off); off += WEL * 2;
  __bf16* Qbf = (__bf16*)(ws + off); off += XEL * 2;
  __bf16* Kbf = (__bf16*)(ws + off); off += XEL * 2;
  __bf16* Vbf = (__bf16*)(ws + off); off += XEL * 2;
  __bf16* Vtb = (__bf16*)(ws + off); off += XEL * 2;
  __bf16* Obf = (__bf16*)(ws + off); off += XEL * 2;

  float* out_main = (float*)d_out;           // [B][N][DIM]
  float* attn     = out_main + XEL;          // [B][H][N][N], also dots scratch

  // 1) conversions
  cvt_bf16_kernel<<<dim3((unsigned)((XEL + 255) / 256)), dim3(256), 0, stream>>>(x, Xbf, (int)XEL);
  cvt_transpose_kernel<<<dim3(DDIM / 256, DDIM), dim3(256), 0, stream>>>(Wq, Wqt, DDIM, DDIM);
  cvt_transpose_kernel<<<dim3(DDIM / 256, DDIM), dim3(256), 0, stream>>>(Wk, Wkt, DDIM, DDIM);
  cvt_transpose_kernel<<<dim3(DDIM / 256, DDIM), dim3(256), 0, stream>>>(Wv, Wvt, DDIM, DDIM);
  cvt_transpose_kernel<<<dim3(DDIM / 256, DDIM), dim3(256), 0, stream>>>(Wo, Wot, DDIM, DDIM);

  // 2) Q/K/V projections (WMMA), scatter to [b][h][n][dh]
  dim3 pg(DDIM / 128, (unsigned)(M / 64));
  proj_gemm_kernel<<<pg, dim3(256), 0, stream>>>(Xbf, Wqt, Qbf);
  proj_gemm_kernel<<<pg, dim3(256), 0, stream>>>(Xbf, Wkt, Kbf);
  proj_gemm_kernel<<<pg, dim3(256), 0, stream>>>(Xbf, Wvt, Vbf);

  // 3) partial RoPE on q, k, v
  long rthreads = (long)BB * HEADS * NNN * (ROT / 2);
  dim3 rg((unsigned)((rthreads + 255) / 256));
  rope_kernel<<<rg, dim3(256), 0, stream>>>(Qbf, rope);
  rope_kernel<<<rg, dim3(256), 0, stream>>>(Kbf, rope);
  rope_kernel<<<rg, dim3(256), 0, stream>>>(Vbf, rope);

  // 3b) transpose V (post-RoPE) for the PV GEMM fast path
  v_transpose_kernel<<<dim3(NNN / 256, DH, BB * HEADS), dim3(256), 0, stream>>>(Vbf, Vtb);

  // 4) dots = scale * Q K^T  -> attn region of d_out
  dots_kernel<<<dim3(NNN / 128, NNN / 64, BB * HEADS), dim3(256), 0, stream>>>(Qbf, Kbf, attn);

  // 5) softmax rows in place (this is the attn output)
  softmax_kernel<<<dim3((unsigned)(BB * HEADS * NNN)), dim3(256), 0, stream>>>(attn);

  // 6) O = P V  -> bf16 [b*n][dim]
  pv_kernel<<<dim3(1, NNN / 128, BB * HEADS), dim3(256), 0, stream>>>(attn, Vtb, Obf);

  // 7) out = O @ Wo + bo + x
  out_gemm_kernel<<<pg, dim3(256), 0, stream>>>(Obf, Wot, bo, x, out_main);

  (void)in_sizes; (void)n_in; (void)out_size; (void)ws_size;
}
